// ModelvD_31903017074984
// MI455X (gfx1250) — compile-verified, matrix operand order checked by reading, not run on hardware
//
#include <hip/hip_runtime.h>
#include <hip/hip_bf16.h>

typedef __attribute__((ext_vector_type(16))) _Float16 v16h;
typedef __attribute__((ext_vector_type(8)))  float    v8f;

#define GB  2
#define GN  2048
#define GD  256
#define GH  8
#define GHID 64
#define GEN 2
#define GALPHA 0.2f
#define GNEG  (-9000000000000000.0f)

// ---- optional CDNA5 async global->LDS path (guarded; falls back cleanly) ----
#if defined(__HIP_DEVICE_COMPILE__) && defined(__gfx1250__)
#if __has_builtin(__builtin_amdgcn_global_load_async_to_lds_b128) && \
    __has_builtin(__builtin_amdgcn_s_wait_asynccnt)
#define HAVE_ASYNC_LDS 1
#endif
#endif

#if defined(HAVE_ASYNC_LDS)
typedef int v4i __attribute__((ext_vector_type(4)));
typedef __attribute__((address_space(1))) v4i as1_v4i;
typedef __attribute__((address_space(3))) v4i as3_v4i;
__device__ __forceinline__ void async_copy16(const void* gsrc, void* lds) {
    __builtin_amdgcn_global_load_async_to_lds_b128(
        (as1_v4i*)(unsigned long long)(size_t)gsrc,
        (as3_v4i*)(unsigned)(size_t)lds, 0, 0);
}
__device__ __forceinline__ void async_wait0() {
    __builtin_amdgcn_s_wait_asynccnt(0);
}
#endif

union H8 { _Float16 h[8]; uint4 u; };

__device__ __forceinline__ float masked_val(float er, float e2v, float av) {
    float e = er + e2v;
    e = (e >= 0.f) ? e : GALPHA * e;
    float v = (av != 0.f) ? e * av : 0.f;
    return (v == 0.f) ? GNEG : v;
}

// ---------------------------------------------------------------------------
// Phase 0: pack W_att (H,D,HID) f32 -> WMMA B-fragment order, f16.
// WB layout: [h][kt(8)][g(4)][lane(32)][16 halfwords]  (32B contiguous/lane)
// ---------------------------------------------------------------------------
__global__ void gat_packW_kernel(const float* __restrict__ Watt,
                                 _Float16* __restrict__ WB) {
    int idx = blockIdx.x * blockDim.x + threadIdx.x;
    if (idx >= GH * 8 * 4 * 32) return;
    int lane = idx & 31;
    int g    = (idx >> 5) & 3;
    int kt   = (idx >> 7) & 7;
    int hh   = idx >> 10;
    int kbase = (lane >> 4) * 8;
    int n     = lane & 15;
    v16h hb;
#pragma unroll
    for (int tt = 0; tt < 16; ++tt) {
        int k = kt * 32 + kbase + tt + ((tt >= 8) ? 8 : 0);
        hb[tt] = (_Float16)Watt[((size_t)hh * GD + k) * GHID + g * 16 + n];
    }
    *(v16h*)(WB + (size_t)idx * 16) = hb;
}

// ---------------------------------------------------------------------------
// Phase 1: wh = fea @ W_att (16x64 tile per wave32, f16 WMMA, f32 accum).
// Emits wh32 (row-major, for e1/e2) and Bfrag (B-fragment-swizzled f16 copy
// consumed coalesced by phase 2).
// ---------------------------------------------------------------------------
__global__ __launch_bounds__(32) void gat_wh_kernel(const float* __restrict__ fea,
                                                    const _Float16* __restrict__ WB,
                                                    float* __restrict__ wh32,
                                                    _Float16* __restrict__ Bfrag) {
    const int TILES = GN / 16;
    int blk = blockIdx.x;
    int t  = blk % TILES;
    int hh = (blk / TILES) % GH;
    int b  = blk / (TILES * GH);
    int i0 = t * 16;

    int lane  = threadIdx.x;
    int m     = lane & 15;
    int hsel  = lane >> 4;
    int kbase = hsel * 8;
    int ncol  = lane & 15;

    v8f c0 = {}, c1 = {}, c2 = {}, c3 = {};
    const float* feaRow = fea + ((size_t)b * GN + (i0 + m)) * GD;

    for (int kt = 0; kt < GD / 32; ++kt) {
        int k0 = kt * 32;
        float4 f0 = *(const float4*)&feaRow[k0 + kbase];
        float4 f1 = *(const float4*)&feaRow[k0 + kbase + 4];
        float4 f2 = *(const float4*)&feaRow[k0 + kbase + 16];
        float4 f3 = *(const float4*)&feaRow[k0 + kbase + 20];
        v16h a;
        a[0]=(_Float16)f0.x; a[1]=(_Float16)f0.y; a[2]=(_Float16)f0.z; a[3]=(_Float16)f0.w;
        a[4]=(_Float16)f1.x; a[5]=(_Float16)f1.y; a[6]=(_Float16)f1.z; a[7]=(_Float16)f1.w;
        a[8]=(_Float16)f2.x; a[9]=(_Float16)f2.y; a[10]=(_Float16)f2.z; a[11]=(_Float16)f2.w;
        a[12]=(_Float16)f3.x; a[13]=(_Float16)f3.y; a[14]=(_Float16)f3.z; a[15]=(_Float16)f3.w;

        size_t wb = (((size_t)hh * 8 + kt) * 4) * 32 * 16;
        v16h b0f = *(const v16h*)(WB + wb + (size_t)(0 * 32 + lane) * 16);
        v16h b1f = *(const v16h*)(WB + wb + (size_t)(1 * 32 + lane) * 16);
        v16h b2f = *(const v16h*)(WB + wb + (size_t)(2 * 32 + lane) * 16);
        v16h b3f = *(const v16h*)(WB + wb + (size_t)(3 * 32 + lane) * 16);

        c0 = __builtin_amdgcn_wmma_f32_16x16x32_f16(false, a, false, b0f, (short)0, c0, false, false);
        c1 = __builtin_amdgcn_wmma_f32_16x16x32_f16(false, a, false, b1f, (short)0, c1, false, false);
        c2 = __builtin_amdgcn_wmma_f32_16x16x32_f16(false, a, false, b2f, (short)0, c2, false, false);
        c3 = __builtin_amdgcn_wmma_f32_16x16x32_f16(false, a, false, b3f, (short)0, c3, false, false);
    }

    int rowadd = hsel * 8;
#pragma unroll
    for (int r = 0; r < 8; ++r) {
        int row = rowadd + r;
        size_t base = (((size_t)b * GH + hh) * GN + (i0 + row)) * GHID + ncol;
        wh32[base +  0] = c0[r];
        wh32[base + 16] = c1[r];
        wh32[base + 32] = c2[r];
        wh32[base + 48] = c3[r];
    }

    int jt = t >> 1, hf = t & 1;
    size_t fbase = (((size_t)(b * GH + hh) * 64 + jt) * 4) * 32 * 16;
    H8 p0, p1, p2, p3;
#pragma unroll
    for (int r = 0; r < 8; ++r) {
        p0.h[r] = (_Float16)c0[r];
        p1.h[r] = (_Float16)c1[r];
        p2.h[r] = (_Float16)c2[r];
        p3.h[r] = (_Float16)c3[r];
    }
    *(uint4*)(Bfrag + fbase + (size_t)(0 * 32 + lane) * 16 + hf * 8) = p0.u;
    *(uint4*)(Bfrag + fbase + (size_t)(1 * 32 + lane) * 16 + hf * 8) = p1.u;
    *(uint4*)(Bfrag + fbase + (size_t)(2 * 32 + lane) * 16 + hf * 8) = p2.u;
    *(uint4*)(Bfrag + fbase + (size_t)(3 * 32 + lane) * 16 + hf * 8) = p3.u;
}

// ---------------------------------------------------------------------------
// Phase 1b: e1/e2 per (b,h,n)
// ---------------------------------------------------------------------------
__global__ void gat_e_kernel(const float* __restrict__ wh32,
                             const float* __restrict__ a_att,
                             float* __restrict__ e1, float* __restrict__ e2) {
    int idx = blockIdx.x * blockDim.x + threadIdx.x;
    if (idx >= GB * GH * GN) return;
    int hh = (idx / GN) % GH;
    const float* w  = wh32 + (size_t)idx * GHID;
    const float* a1 = a_att + (size_t)hh * 2 * GHID;
    const float* a2 = a1 + GHID;
    float s1 = 0.f, s2 = 0.f;
    for (int o = 0; o < GHID; o += 4) {
        float4 wv = *(const float4*)&w[o];
        s1 += wv.x * a1[o] + wv.y * a1[o + 1] + wv.z * a1[o + 2] + wv.w * a1[o + 3];
        s2 += wv.x * a2[o] + wv.y * a2[o + 1] + wv.z * a2[o + 2] + wv.w * a2[o + 3];
    }
    e1[idx] = s1; e2[idx] = s2;
}

// ---------------------------------------------------------------------------
// Phase 2: fused masked softmax + (att @ wh) WMMA + ELU -> x (B,N,H*HID).
// e2 row staged into LDS via async global->LDS when available.
// ---------------------------------------------------------------------------
__global__ __launch_bounds__(32) void gat_attn_kernel(const float* __restrict__ adj,
                                                      const float* __restrict__ e1g,
                                                      const float* __restrict__ e2g,
                                                      const _Float16* __restrict__ Bfrag,
                                                      float* __restrict__ xbuf) {
    __shared__ float e2s[GN];
    __shared__ float e1s[16];

    const int TILES = GN / 16;
    int blk = blockIdx.x;
    int t  = blk % TILES;
    int hh = (blk / TILES) % GH;
    int b  = blk / (TILES * GH);
    int i0 = t * 16;

    int lane  = threadIdx.x;
    int r     = lane & 15;
    int hsel  = lane >> 4;
    int kbase = hsel * 8;

    size_t ebase = ((size_t)b * GH + hh) * GN;
#if defined(HAVE_ASYNC_LDS)
    {
        const char* gsrc = (const char*)(e2g + ebase);
        char* ldst = (char*)e2s;
#pragma unroll
        for (int it = 0; it < (GN * 4) / (32 * 16); ++it) {
            int off = (it * 32 + lane) * 16;
            async_copy16(gsrc + off, ldst + off);
        }
    }
#else
    for (int j = lane * 4; j < GN; j += 32 * 4)
        *(float4*)&e2s[j] = *(const float4*)&e2g[ebase + j];
#endif
    if (lane < 16) e1s[lane] = e1g[ebase + i0 + lane];
#if defined(HAVE_ASYNC_LDS)
    async_wait0();
#endif
    __syncthreads();

    const float* adjRow = adj + ((size_t)b * GN + (i0 + r)) * GN;
    float er = e1s[r];

    // ---- pass A: row max ----
    float mx = GNEG;
    int jlo = hsel * (GN / 2), jhi = jlo + (GN / 2);
    for (int j = jlo; j < jhi; j += 4) {
        __builtin_prefetch(&adjRow[j + 256], 0, 0);   // global_prefetch_b8
        float4 a4 = *(const float4*)&adjRow[j];
        float4 e4 = *(const float4*)&e2s[j];
        mx = fmaxf(mx, masked_val(er, e4.x, a4.x));
        mx = fmaxf(mx, masked_val(er, e4.y, a4.y));
        mx = fmaxf(mx, masked_val(er, e4.z, a4.z));
        mx = fmaxf(mx, masked_val(er, e4.w, a4.w));
    }
    mx = fmaxf(mx, __shfl_xor(mx, 16, 32));

    // ---- pass B: sum of exp ----
    float s = 0.f;
    for (int j = jlo; j < jhi; j += 4) {
        float4 a4 = *(const float4*)&adjRow[j];
        float4 e4 = *(const float4*)&e2s[j];
        s += __expf(masked_val(er, e4.x, a4.x) - mx);
        s += __expf(masked_val(er, e4.y, a4.y) - mx);
        s += __expf(masked_val(er, e4.z, a4.z) - mx);
        s += __expf(masked_val(er, e4.w, a4.w) - mx);
    }
    s += __shfl_xor(s, 16, 32);
    float inv_s = 1.f / s;

    // ---- pass C: normalized att tiles -> WMMA A-frags, accumulate att@wh ----
    v8f c0 = {}, c1 = {}, c2 = {}, c3 = {};
    size_t bh64 = (size_t)(b * GH + hh) * 64;
    for (int j0 = 0; j0 < GN; j0 += 32) {
        float4 q0 = *(const float4*)&adjRow[j0 + kbase];
        float4 q1 = *(const float4*)&adjRow[j0 + kbase + 4];
        float4 q2 = *(const float4*)&adjRow[j0 + kbase + 16];
        float4 q3 = *(const float4*)&adjRow[j0 + kbase + 20];
        float4 z0 = *(const float4*)&e2s[j0 + kbase];
        float4 z1 = *(const float4*)&e2s[j0 + kbase + 4];
        float4 z2 = *(const float4*)&e2s[j0 + kbase + 16];
        float4 z3 = *(const float4*)&e2s[j0 + kbase + 20];
        float avv[16] = {q0.x,q0.y,q0.z,q0.w, q1.x,q1.y,q1.z,q1.w,
                         q2.x,q2.y,q2.z,q2.w, q3.x,q3.y,q3.z,q3.w};
        float evv[16] = {z0.x,z0.y,z0.z,z0.w, z1.x,z1.y,z1.z,z1.w,
                         z2.x,z2.y,z2.z,z2.w, z3.x,z3.y,z3.z,z3.w};
        v16h a;
#pragma unroll
        for (int tt = 0; tt < 16; ++tt)
            a[tt] = (_Float16)(__expf(masked_val(er, evv[tt], avv[tt]) - mx) * inv_s);

        size_t fb = ((bh64 + (size_t)(j0 >> 5)) * 4) * 32 * 16;
        v16h b0f = *(const v16h*)(Bfrag + fb + (size_t)(0 * 32 + lane) * 16);
        v16h b1f = *(const v16h*)(Bfrag + fb + (size_t)(1 * 32 + lane) * 16);
        v16h b2f = *(const v16h*)(Bfrag + fb + (size_t)(2 * 32 + lane) * 16);
        v16h b3f = *(const v16h*)(Bfrag + fb + (size_t)(3 * 32 + lane) * 16);

        c0 = __builtin_amdgcn_wmma_f32_16x16x32_f16(false, a, false, b0f, (short)0, c0, false, false);
        c1 = __builtin_amdgcn_wmma_f32_16x16x32_f16(false, a, false, b1f, (short)0, c1, false, false);
        c2 = __builtin_amdgcn_wmma_f32_16x16x32_f16(false, a, false, b2f, (short)0, c2, false, false);
        c3 = __builtin_amdgcn_wmma_f32_16x16x32_f16(false, a, false, b3f, (short)0, c3, false, false);
    }

    // ---- epilogue: ELU, store transposed as x (B, N, H*HID) ----
    int rowadd = hsel * 8;
    int ncol = lane & 15;
#pragma unroll
    for (int rr = 0; rr < 8; ++rr) {
        int row = rowadd + rr;
        size_t xb = (((size_t)b * GN) + i0 + row) * (GH * GHID) + (size_t)hh * GHID + ncol;
        float v;
        v = c0[rr]; xbuf[xb +  0] = (v > 0.f) ? v : (__expf(v) - 1.f);
        v = c1[rr]; xbuf[xb + 16] = (v > 0.f) ? v : (__expf(v) - 1.f);
        v = c2[rr]; xbuf[xb + 32] = (v > 0.f) ? v : (__expf(v) - 1.f);
        v = c3[rr]; xbuf[xb + 48] = (v > 0.f) ? v : (__expf(v) - 1.f);
    }
}

// ---------------------------------------------------------------------------
// Phase 3: wh2 = x @ W_last (512x2), plus eb1/eb2 scalars
// ---------------------------------------------------------------------------
__global__ void gat_wh2_kernel(const float* __restrict__ xbuf,
                               const float* __restrict__ Wlast,
                               const float* __restrict__ alast,
                               float* __restrict__ wh2,
                               float* __restrict__ eb1, float* __restrict__ eb2) {
    int idx = blockIdx.x * blockDim.x + threadIdx.x;
    if (idx >= GB * GN) return;
    const float* xr = xbuf + (size_t)idx * (GH * GHID);
    float s0 = 0.f, s1 = 0.f;
    for (int c = 0; c < GH * GHID; c += 4) {
        float4 xv = *(const float4*)&xr[c];
        s0 += xv.x * Wlast[(c + 0) * GEN] + xv.y * Wlast[(c + 1) * GEN]
            + xv.z * Wlast[(c + 2) * GEN] + xv.w * Wlast[(c + 3) * GEN];
        s1 += xv.x * Wlast[(c + 0) * GEN + 1] + xv.y * Wlast[(c + 1) * GEN + 1]
            + xv.z * Wlast[(c + 2) * GEN + 1] + xv.w * Wlast[(c + 3) * GEN + 1];
    }
    wh2[idx * 2 + 0] = s0;
    wh2[idx * 2 + 1] = s1;
    eb1[idx] = s0 * alast[0] + s1 * alast[1];
    eb2[idx] = s0 * alast[2] + s1 * alast[3];
}

// ---------------------------------------------------------------------------
// Phase 4: fused stage-2 masked softmax + (att @ wh2) + ELU -> out (B,N,2).
// adj row staged ONCE into LDS (async global->LDS when available), then both
// the max pass and the exp/accumulate pass read LDS: halves adj HBM traffic.
// ---------------------------------------------------------------------------
__global__ __launch_bounds__(256) void gat_out_kernel(const float* __restrict__ adj,
                                                      const float* __restrict__ wh2,
                                                      const float* __restrict__ eb1,
                                                      const float* __restrict__ eb2,
                                                      float* __restrict__ out) {
    __shared__ float adjs[GN];
    __shared__ float red[256];
    int blk = blockIdx.x;
    int b = blk / GN;
    int i = blk % GN;
    int tid = threadIdx.x;

    const float* adjRow = adj + ((size_t)b * GN + i) * GN;
    const float* e2r = eb2 + (size_t)b * GN;
    const float2* w2 = (const float2*)(wh2 + (size_t)b * GN * 2);
    float ei = eb1[(size_t)b * GN + i];

#if defined(HAVE_ASYNC_LDS)
    async_copy16((const char*)adjRow + tid * 32,      (char*)adjs + tid * 32);
    async_copy16((const char*)adjRow + tid * 32 + 16, (char*)adjs + tid * 32 + 16);
    async_wait0();
#else
    for (int j = tid * 4; j < GN; j += 256 * 4)
        *(float4*)&adjs[j] = *(const float4*)&adjRow[j];
#endif
    __syncthreads();

    float mx = GNEG;
    for (int j = tid * 4; j < GN; j += 256 * 4) {
        float4 a4 = *(const float4*)&adjs[j];
        float4 e4 = *(const float4*)&e2r[j];
        mx = fmaxf(mx, masked_val(ei, e4.x, a4.x));
        mx = fmaxf(mx, masked_val(ei, e4.y, a4.y));
        mx = fmaxf(mx, masked_val(ei, e4.z, a4.z));
        mx = fmaxf(mx, masked_val(ei, e4.w, a4.w));
    }
    red[tid] = mx; __syncthreads();
    for (int st = 128; st > 0; st >>= 1) { if (tid < st) red[tid] = fmaxf(red[tid], red[tid + st]); __syncthreads(); }
    mx = red[0]; __syncthreads();

    float s = 0.f, o0 = 0.f, o1 = 0.f;
    for (int j = tid * 4; j < GN; j += 256 * 4) {
        float4 a4 = *(const float4*)&adjs[j];
        float4 e4 = *(const float4*)&e2r[j];
        float p;
        float2 wv;
        p = __expf(masked_val(ei, e4.x, a4.x) - mx); wv = w2[j + 0]; s += p; o0 += p * wv.x; o1 += p * wv.y;
        p = __expf(masked_val(ei, e4.y, a4.y) - mx); wv = w2[j + 1]; s += p; o0 += p * wv.x; o1 += p * wv.y;
        p = __expf(masked_val(ei, e4.z, a4.z) - mx); wv = w2[j + 2]; s += p; o0 += p * wv.x; o1 += p * wv.y;
        p = __expf(masked_val(ei, e4.w, a4.w) - mx); wv = w2[j + 3]; s += p; o0 += p * wv.x; o1 += p * wv.y;
    }
    red[tid] = s;  __syncthreads();
    for (int st = 128; st > 0; st >>= 1) { if (tid < st) red[tid] += red[tid + st]; __syncthreads(); }
    s = red[0]; __syncthreads();
    red[tid] = o0; __syncthreads();
    for (int st = 128; st > 0; st >>= 1) { if (tid < st) red[tid] += red[tid + st]; __syncthreads(); }
    o0 = red[0]; __syncthreads();
    red[tid] = o1; __syncthreads();
    for (int st = 128; st > 0; st >>= 1) { if (tid < st) red[tid] += red[tid + st]; __syncthreads(); }
    o1 = red[0];

    if (tid == 0) {
        float v0 = o0 / s, v1 = o1 / s;
        out[((size_t)b * GN + i) * 2 + 0] = (v0 > 0.f) ? v0 : (__expf(v0) - 1.f);
        out[((size_t)b * GN + i) * 2 + 1] = (v1 > 0.f) ? v1 : (__expf(v1) - 1.f);
    }
}

// ---------------------------------------------------------------------------
extern "C" void kernel_launch(void* const* d_in, const int* in_sizes, int n_in,
                              void* d_out, int out_size, void* d_ws, size_t ws_size,
                              hipStream_t stream) {
    const float* fea   = (const float*)d_in[0];   // (B,N,D)
    const float* adj   = (const float*)d_in[1];   // (B,N,N)
    const float* Watt  = (const float*)d_in[2];   // (H,D,HID)
    const float* a_att = (const float*)d_in[3];   // (H,2*HID,1)
    const float* Wlast = (const float*)d_in[4];   // (H*HID,EN)
    const float* alast = (const float*)d_in[5];   // (2*EN,1)
    float* out = (float*)d_out;                   // (B,N,EN)

    char* ws = (char*)d_ws;
    size_t off = 0;
    float*    wh32  = (float*)(ws + off);    off += (size_t)GB * GH * GN * GHID * sizeof(float);
    _Float16* Bfrag = (_Float16*)(ws + off); off += (size_t)GB * GH * 64 * 4 * 32 * 16 * sizeof(_Float16);
    _Float16* WB    = (_Float16*)(ws + off); off += (size_t)GH * 8 * 4 * 32 * 16 * sizeof(_Float16);
    float*    e1    = (float*)(ws + off);    off += (size_t)GB * GH * GN * sizeof(float);
    float*    e2    = (float*)(ws + off);    off += (size_t)GB * GH * GN * sizeof(float);
    float*    xbuf  = (float*)(ws + off);    off += (size_t)GB * GN * GH * GHID * sizeof(float);
    float*    wh2   = (float*)(ws + off);    off += (size_t)GB * GN * GEN * sizeof(float);
    float*    eb1   = (float*)(ws + off);    off += (size_t)GB * GN * sizeof(float);
    float*    eb2   = (float*)(ws + off);    off += (size_t)GB * GN * sizeof(float);
    (void)ws_size; (void)in_sizes; (void)n_in; (void)out_size;

    const int TILES = GN / 16;

    gat_packW_kernel<<<(GH * 8 * 4 * 32 + 255) / 256, 256, 0, stream>>>(Watt, WB);
    gat_wh_kernel<<<GB * GH * TILES, 32, 0, stream>>>(fea, WB, wh32, Bfrag);
    gat_e_kernel<<<(GB * GH * GN + 255) / 256, 256, 0, stream>>>(wh32, a_att, e1, e2);
    gat_attn_kernel<<<GB * GH * TILES, 32, 0, stream>>>(adj, e1, e2, Bfrag, xbuf);
    gat_wh2_kernel<<<(GB * GN + 255) / 256, 256, 0, stream>>>(xbuf, Wlast, alast, wh2, eb1, eb2);
    gat_out_kernel<<<GB * GN, 256, 0, stream>>>(adj, wh2, eb1, eb2, out);
}
